// BertAttention_43997644981001
// MI455X (gfx1250) — matrix operand order repaired and checked by reading, not compiled, not run on hardware
//
#include <hip/hip_runtime.h>

#define B_  8
#define S_  1024
#define D_  768
#define H_  12
#define HD_ 64
#define M_  (B_*S_)   // 8192

typedef __bf16 bf16_t;
typedef __bf16 bf16x4  __attribute__((ext_vector_type(4)));
typedef __bf16 bf16x8  __attribute__((ext_vector_type(8)));
typedef __bf16 bf16x16 __attribute__((ext_vector_type(16)));
typedef float  floatx8 __attribute__((ext_vector_type(8)));
typedef int    v4i     __attribute__((ext_vector_type(4)));

__device__ __forceinline__ bf16_t to_bf(float x) { return (bf16_t)x; }

__device__ __forceinline__ floatx8 zero8() {
  floatx8 z = {0.f,0.f,0.f,0.f,0.f,0.f,0.f,0.f};
  return z;
}

// ---- CDNA5 async global->LDS staging (ASYNCcnt), with safe fallback -------
#if defined(__HIP_DEVICE_COMPILE__) && __has_builtin(__builtin_amdgcn_global_load_async_to_lds_b128) && __has_builtin(__builtin_amdgcn_s_wait_asynccnt)
#define HAVE_ASYNC 1
#else
#define HAVE_ASYNC 0
#endif

#if HAVE_ASYNC
typedef __attribute__((address_space(1))) v4i* as1_v4i_ptr;
typedef __attribute__((address_space(3))) v4i* as3_v4i_ptr;
#endif

__device__ __forceinline__ void stage16(const bf16_t* g, bf16_t* l) {
#if HAVE_ASYNC
  __builtin_amdgcn_global_load_async_to_lds_b128(
      (as1_v4i_ptr)(g), (as3_v4i_ptr)(l), 0, 0);
#else
  *(uint4*)l = *(const uint4*)g;
#endif
}

__device__ __forceinline__ void stage_fence() {
#if HAVE_ASYNC
  __builtin_amdgcn_s_wait_asynccnt(0);
#endif
}

// A-matrix fragment (16xK=32 bf16), ISA 7.12.2: K chunks [koff,koff+8) and
// [koff+16,koff+24) with koff = 8*(lane>=16).
__device__ __forceinline__ bf16x16 frag_a(const bf16_t* base, int row, int stride, int lane) {
  const bf16_t* p = base + row * stride + ((lane >> 4) << 3);
  bf16x8 lo = *(const bf16x8*)(p);
  bf16x8 hi = *(const bf16x8*)(p + 16);
  bf16x16 f;
#pragma unroll
  for (int i = 0; i < 8; ++i) { f[i] = lo[i]; f[i + 8] = hi[i]; }
  return f;
}

// B-matrix fragment (K=32 x 16 bf16): lane holds column n = lane&15,
// K chunks [koff,koff+8) and [koff+16,koff+24), koff = 8*(lane>=16). LDS layout is [n][k].
__device__ __forceinline__ bf16x16 frag_b(const bf16_t* base, int row, int stride, int lane) {
  const bf16_t* p = base + row * stride + ((lane >> 4) << 3);
  bf16x8 lo = *(const bf16x8*)(p);
  bf16x8 hi = *(const bf16x8*)(p + 16);
  bf16x16 f;
#pragma unroll
  for (int i = 0; i < 8; ++i) { f[i] = lo[i]; f[i + 8] = hi[i]; }
  return f;
}

__device__ __forceinline__ floatx8 wmma_bf16(bf16x16 a, bf16x16 b, floatx8 c) {
  return __builtin_amdgcn_wmma_f32_16x16x32_bf16(false, a, false, b, (short)0, c, false, false);
}

// ---------------------------------------------------------------------------
// Kernel 0: one-shot f32 -> bf16 conversion (X, Wq, Wk, Wv, Wo).
// ---------------------------------------------------------------------------
__global__ __launch_bounds__(256)
void conv_kernel(const float* __restrict__ src, bf16_t* __restrict__ dst, int n) {
  int i = (blockIdx.x * 256 + threadIdx.x) * 4;
  if (i + 3 < n) {
    float4 v = *(const float4*)(src + i);
    bf16x4 o = { to_bf(v.x), to_bf(v.y), to_bf(v.z), to_bf(v.w) };
    *(bf16x4*)(dst + i) = o;
  }
}

// ---------------------------------------------------------------------------
// Kernel 1: QKV projection GEMM (bf16 in, bf16 out).
// C[m,n] = (sum_k X[m,k]*W[n,k] + bias[n]) * (z==0 ? 0.125 : 1)
// Block: 256 threads (8 waves). Tile 128x128, wave does 32x64 (2x4 wmma).
// grid = (6, 64, 3): x=N tile, y=M tile, z selects Q/K/V.
// ---------------------------------------------------------------------------
__global__ __launch_bounds__(256)
void qkv_gemm_kernel(const bf16_t* __restrict__ Xb,
                     const bf16_t* __restrict__ Wqb, const float* __restrict__ bq,
                     const bf16_t* __restrict__ Wkb, const float* __restrict__ bk,
                     const bf16_t* __restrict__ Wvb, const float* __restrict__ bv,
                     bf16_t* __restrict__ Qo, bf16_t* __restrict__ Ko,
                     bf16_t* __restrict__ Vo) {
  const bf16_t* W; const float* bias; bf16_t* out; float oscale;
  if (blockIdx.z == 0)      { W = Wqb; bias = bq; out = Qo; oscale = 0.125f; }
  else if (blockIdx.z == 1) { W = Wkb; bias = bk; out = Ko; oscale = 1.0f; }
  else                      { W = Wvb; bias = bv; out = Vo; oscale = 1.0f; }

  __shared__ bf16_t As[128 * 32];
  __shared__ bf16_t Bs[128 * 32];

  const int tid = threadIdx.x;
  const int lane = tid & 31, wave = tid >> 5;
  const int wm = wave & 3, wn = wave >> 2;
  const int m0 = blockIdx.y * 128, n0 = blockIdx.x * 128;

  floatx8 acc[2][4];
#pragma unroll
  for (int i = 0; i < 2; ++i)
#pragma unroll
    for (int j = 0; j < 4; ++j) acc[i][j] = zero8();

  for (int k0 = 0; k0 < D_; k0 += 32) {
    __syncthreads();
    // 128x32 bf16 tile = 512 16B chunks; 2 per thread, for A and for B.
#pragma unroll
    for (int i = 0; i < 2; ++i) {
      int ch = tid + i * 256;
      int r = ch >> 2, c8 = (ch & 3) << 3;
      stage16(&Xb[(size_t)(m0 + r) * D_ + k0 + c8], &As[r * 32 + c8]);
      stage16(&W [(size_t)(n0 + r) * D_ + k0 + c8], &Bs[r * 32 + c8]);
    }
    stage_fence();
    __syncthreads();

    bf16x16 a[2], b[4];
#pragma unroll
    for (int i = 0; i < 2; ++i) a[i] = frag_a(As, wm * 32 + i * 16 + (lane & 15), 32, lane);
#pragma unroll
    for (int j = 0; j < 4; ++j) b[j] = frag_b(Bs, wn * 64 + j * 16 + (lane & 15), 32, lane);
#pragma unroll
    for (int i = 0; i < 2; ++i)
#pragma unroll
      for (int j = 0; j < 4; ++j) acc[i][j] = wmma_bf16(a[i], b[j], acc[i][j]);
  }

  const int rhalf = (lane >> 4) << 3;
#pragma unroll
  for (int i = 0; i < 2; ++i)
#pragma unroll
    for (int j = 0; j < 4; ++j) {
      int c = n0 + wn * 64 + j * 16 + (lane & 15);
      float bb = bias[c];
#pragma unroll
      for (int e = 0; e < 8; ++e) {
        int r = m0 + wm * 32 + i * 16 + rhalf + e;
        out[(size_t)r * D_ + c] = to_bf((acc[i][j][e] + bb) * oscale);
      }
    }
}

// ---------------------------------------------------------------------------
// Kernel 2: flash-style attention per (b, h, 128 query rows).
// Q pre-scaled by 1/8 in kernel 1; Q/K staged async; V transposed (bit-move).
// grid = (8, 12, 8): x=q tile, y=head, z=batch.
// ---------------------------------------------------------------------------
__global__ __launch_bounds__(256)
void attn_kernel(const bf16_t* __restrict__ Qb, const bf16_t* __restrict__ Kb,
                 const bf16_t* __restrict__ Vb, const float* __restrict__ mask,
                 bf16_t* __restrict__ Cb) {
  __shared__ bf16_t Qs[128 * 64];
  __shared__ bf16_t Ks[32 * 64];
  __shared__ bf16_t Vt[64 * 32];
  __shared__ bf16_t Ps[8][16 * 32];

  const int tid = threadIdx.x, lane = tid & 31, wave = tid >> 5;
  const int h = blockIdx.y, b = blockIdx.z;
  const int q0 = blockIdx.x * 128;
  const int colbase = h * HD_;
  const int rowbase = b * S_;

  // Stage Q: 128x64 bf16 = 1024 16B chunks, 4 per thread (async).
#pragma unroll
  for (int i = 0; i < 4; ++i) {
    int ch = tid + i * 256;
    int r = ch >> 3, c8 = (ch & 7) << 3;
    stage16(&Qb[(size_t)(rowbase + q0 + r) * D_ + colbase + c8], &Qs[r * 64 + c8]);
  }
  stage_fence();
  __syncthreads();

  bf16x16 qa[2];
  qa[0] = frag_a(Qs,      wave * 16 + (lane & 15), 64, lane);
  qa[1] = frag_a(Qs + 32, wave * 16 + (lane & 15), 64, lane);

  float m_r[8], l_r[8];
  floatx8 ctx[4];
#pragma unroll
  for (int i = 0; i < 8; ++i) { m_r[i] = -1e30f; l_r[i] = 0.f; }
#pragma unroll
  for (int t = 0; t < 4; ++t) ctx[t] = zero8();

  const int rhalf = (lane >> 4) << 3;

  for (int kc = 0; kc < S_; kc += 32) {
    __syncthreads();
    {  // K tile: 32x64 bf16 = 256 16B chunks, 1 per thread (async).
      int r = tid >> 3, c8 = (tid & 7) << 3;
      stage16(&Kb[(size_t)(rowbase + kc + r) * D_ + colbase + c8], &Ks[r * 64 + c8]);
    }
#pragma unroll
    for (int i = 0; i < 8; ++i) {  // V tile transposed: raw bf16 bit-move
      int idx = tid + i * 256;     // 2048 elems
      int r = idx >> 6, d = idx & 63;
      Vt[d * 32 + r] = Vb[(size_t)(rowbase + kc + r) * D_ + colbase + d];
    }
    stage_fence();
    __syncthreads();

    // Scores: two 16x16 key tiles, K-depth 64 in two 32-steps.
    floatx8 s0 = zero8(), s1 = zero8();
    bf16x16 kb;
    kb = frag_b(Ks,      (lane & 15),      64, lane); s0 = wmma_bf16(qa[0], kb, s0);
    kb = frag_b(Ks + 32, (lane & 15),      64, lane); s0 = wmma_bf16(qa[1], kb, s0);
    kb = frag_b(Ks,      16 + (lane & 15), 64, lane); s1 = wmma_bf16(qa[0], kb, s1);
    kb = frag_b(Ks + 32, 16 + (lane & 15), 64, lane); s1 = wmma_bf16(qa[1], kb, s1);

    float mk0 = mask[b * S_ + kc + (lane & 15)];
    float mk1 = mask[b * S_ + kc + 16 + (lane & 15)];

#pragma unroll
    for (int i = 0; i < 8; ++i) {
      float a0 = s0[i] + mk0, a1 = s1[i] + mk1;
      float t = fmaxf(a0, a1);
#pragma unroll
      for (int off = 1; off <= 8; off <<= 1) t = fmaxf(t, __shfl_xor(t, off, 32));
      float nm = fmaxf(m_r[i], t);
      float scale = __expf(m_r[i] - nm);
      float p0 = __expf(a0 - nm), p1 = __expf(a1 - nm);
      float rs = p0 + p1;
#pragma unroll
      for (int off = 1; off <= 8; off <<= 1) rs += __shfl_xor(rs, off, 32);
      l_r[i] = l_r[i] * scale + rs;
      m_r[i] = nm;
#pragma unroll
      for (int t4 = 0; t4 < 4; ++t4) ctx[t4][i] *= scale;
      int prow = i + rhalf;
      Ps[wave][prow * 32 + (lane & 15)]      = to_bf(p0);
      Ps[wave][prow * 32 + 16 + (lane & 15)] = to_bf(p1);
    }

    // PV: ctx += P(16x32) x V(32x64)
    bf16x16 pf = frag_a(&Ps[wave][0], (lane & 15), 32, lane);
#pragma unroll
    for (int t4 = 0; t4 < 4; ++t4) {
      bf16x16 vb = frag_b(Vt, t4 * 16 + (lane & 15), 32, lane);
      ctx[t4] = wmma_bf16(pf, vb, ctx[t4]);
    }
  }

#pragma unroll
  for (int i = 0; i < 8; ++i) {
    float inv = 1.f / l_r[i];
    int q = q0 + wave * 16 + i + rhalf;
#pragma unroll
    for (int t4 = 0; t4 < 4; ++t4) {
      int d = colbase + t4 * 16 + (lane & 15);
      Cb[(size_t)(rowbase + q) * D_ + d] = to_bf(ctx[t4][i] * inv);
    }
  }
}

// ---------------------------------------------------------------------------
// Kernel 3: output projection (head_mask folded into A) + bias + residual
// + LayerNorm. Block owns 16 rows x all 768 cols; B tile streamed async from
// the bf16 copy of Wo; B LDS reused as f32 row buffer for LN. grid = (512).
// ---------------------------------------------------------------------------
__global__ __launch_bounds__(256)
void proj_ln_kernel(const bf16_t* __restrict__ Cb, const float* __restrict__ hidden,
                    const float* __restrict__ head_mask,
                    const bf16_t* __restrict__ Wob, const float* __restrict__ bo,
                    const float* __restrict__ gamma, const float* __restrict__ beta,
                    float* __restrict__ out) {
  __shared__ __align__(16) unsigned char smem[16 * 32 * 2 + 768 * 32 * 2];
  bf16_t* As = (bf16_t*)smem;
  bf16_t* Bs = (bf16_t*)(smem + 16 * 32 * 2);
  float* rowbuf = (float*)(smem + 16 * 32 * 2);  // aliases Bs after GEMM

  const int tid = threadIdx.x, lane = tid & 31, wave = tid >> 5;
  const int m0 = blockIdx.x * 16;

  floatx8 acc[6];
#pragma unroll
  for (int j = 0; j < 6; ++j) acc[j] = zero8();

  for (int k0 = 0; k0 < D_; k0 += 32) {
    __syncthreads();
    // B: 768x32 bf16 = 3072 16B chunks, 12 per thread (async).
#pragma unroll
    for (int i = 0; i < 12; ++i) {
      int ch = tid + i * 256;
      int r = ch >> 2, c8 = (ch & 3) << 3;
      stage16(&Wob[(size_t)r * D_ + k0 + c8], &Bs[r * 32 + c8]);
    }
    // A: 16x32 = 512 elems, 2 per thread; head_mask folded in.
#pragma unroll
    for (int i = 0; i < 2; ++i) {
      int idx = tid + i * 256;
      int r = idx >> 5, c = idx & 31;
      As[idx] = to_bf((float)Cb[(size_t)(m0 + r) * D_ + k0 + c] * head_mask[(k0 + c) >> 6]);
    }
    stage_fence();
    __syncthreads();

    bf16x16 af = frag_a(As, (lane & 15), 32, lane);
#pragma unroll
    for (int j = 0; j < 6; ++j) {
      bf16x16 bfrag = frag_b(Bs, wave * 96 + j * 16 + (lane & 15), 32, lane);
      acc[j] = wmma_bf16(af, bfrag, acc[j]);
    }
  }

  __syncthreads();
  const int rhalf = (lane >> 4) << 3;
#pragma unroll
  for (int j = 0; j < 6; ++j) {
    int c = wave * 96 + j * 16 + (lane & 15);
    float bb = bo[c];
#pragma unroll
    for (int e = 0; e < 8; ++e) {
      int r = e + rhalf;
      rowbuf[r * D_ + c] = acc[j][e] + bb + hidden[(size_t)(m0 + r) * D_ + c];
    }
  }
  __syncthreads();

  // LayerNorm: wave handles rows 2*wave, 2*wave+1
#pragma unroll
  for (int rr = 0; rr < 2; ++rr) {
    int r = wave * 2 + rr;
    float s = 0.f, s2 = 0.f;
    for (int c = lane; c < D_; c += 32) {
      float x = rowbuf[r * D_ + c];
      s += x; s2 += x * x;
    }
#pragma unroll
    for (int off = 1; off <= 16; off <<= 1) {
      s  += __shfl_xor(s,  off, 32);
      s2 += __shfl_xor(s2, off, 32);
    }
    float mu  = s * (1.f / D_);
    float var = s2 * (1.f / D_) - mu * mu;
    float rsd = rsqrtf(var + 1e-12f);
    for (int c = lane; c < D_; c += 32) {
      float x = rowbuf[r * D_ + c];
      out[(size_t)(m0 + r) * D_ + c] = (x - mu) * rsd * gamma[c] + beta[c];
    }
  }
}

// ---------------------------------------------------------------------------
extern "C" void kernel_launch(void* const* d_in, const int* in_sizes, int n_in,
                              void* d_out, int out_size, void* d_ws, size_t ws_size,
                              hipStream_t stream) {
  (void)in_sizes; (void)n_in; (void)out_size; (void)ws_size;
  const float* hidden    = (const float*)d_in[0];
  const float* attn_mask = (const float*)d_in[1];
  const float* head_mask = (const float*)d_in[2];
  const float* Wq = (const float*)d_in[3];
  const float* bq = (const float*)d_in[4];
  const float* Wk = (const float*)d_in[5];
  const float* bk = (const float*)d_in[6];
  const float* Wv = (const float*)d_in[7];
  const float* bv = (const float*)d_in[8];
  const float* Wo = (const float*)d_in[9];
  const float* bo = (const float*)d_in[10];
  const float* gamma = (const float*)d_in[11];
  const float* beta  = (const float*)d_in[12];
  float* out = (float*)d_out;

  const size_t NACT = (size_t)M_ * D_;   // 6291456
  const size_t NW   = (size_t)D_ * D_;   // 589824
  bf16_t* Xb  = (bf16_t*)d_ws;
  bf16_t* Wqb = Xb  + NACT;
  bf16_t* Wkb = Wqb + NW;
  bf16_t* Wvb = Wkb + NW;
  bf16_t* Wob = Wvb + NW;
  bf16_t* Qb  = Wob + NW;
  bf16_t* Kb  = Qb  + NACT;
  bf16_t* Vb  = Kb  + NACT;
  bf16_t* Cb  = Vb  + NACT;

  conv_kernel<<<dim3((int)(NACT / 1024)), 256, 0, stream>>>(hidden, Xb, (int)NACT);
  conv_kernel<<<dim3((int)(NW / 1024)),   256, 0, stream>>>(Wq, Wqb, (int)NW);
  conv_kernel<<<dim3((int)(NW / 1024)),   256, 0, stream>>>(Wk, Wkb, (int)NW);
  conv_kernel<<<dim3((int)(NW / 1024)),   256, 0, stream>>>(Wv, Wvb, (int)NW);
  conv_kernel<<<dim3((int)(NW / 1024)),   256, 0, stream>>>(Wo, Wob, (int)NW);

  qkv_gemm_kernel<<<dim3(D_ / 128, M_ / 128, 3), 256, 0, stream>>>(
      Xb, Wqb, bq, Wkb, bk, Wvb, bv, Qb, Kb, Vb);

  attn_kernel<<<dim3(S_ / 128, H_, B_), 256, 0, stream>>>(
      Qb, Kb, Vb, attn_mask, Cb);

  proj_ln_kernel<<<dim3(M_ / 16), 256, 0, stream>>>(
      Cb, hidden, head_mask, Wob, bo, gamma, beta, out);
}
